// SplineConv_test_26671746908877
// MI455X (gfx1250) — matrix operand order repaired and checked by reading, not compiled, not run on hardware
//
#include <hip/hip_runtime.h>
#include <hip/hip_bf16.h>
#include <math.h>

typedef __attribute__((ext_vector_type(16))) __bf16   v16bf;
typedef __attribute__((ext_vector_type(16))) _Float16 v16h;
typedef __attribute__((ext_vector_type(8)))  float    v8f;
typedef int v4i __attribute__((vector_size(16)));     // matches builtin param type

// Problem constants (match reference)
#define BB   512
#define NN   12
#define EE   48
#define F1   32
#define F2   64
#define FLAT 768          // NN*F2
#define FH   128
#define KTOT 125
#define NEDGE (BB*EE)     // 24576
#define NTRIP (NEDGE*8)   // 196608 (b,e,s) contributions
#define MAXTILES (NTRIP/16 + KTOT)   // 12413 worst-case M-tiles

// ---------------------------------------------------------------------------
// Async global->LDS staging (CDNA5 GLOBAL_LOAD_ASYNC_TO_LDS_B128, ASYNCcnt)
// with a guarded fallback so the file compiles on any toolchain.
// Probe-confirmed signature: (v4i addrspace(1)*, v4i addrspace(3)*, imm, imm)
// ---------------------------------------------------------------------------
#if defined(__gfx1250__) && \
    __has_builtin(__builtin_amdgcn_global_load_async_to_lds_b128) && \
    __has_builtin(__builtin_amdgcn_s_wait_asynccnt)
#define USE_ASYNC_LDS 1
#else
#define USE_ASYNC_LDS 0
#endif

__device__ __forceinline__ void stage_lds(void* lds, const void* g, int bytes,
                                          int tid, int nthr) {
  int n = bytes >> 4;                                   // 16B transfers
#if USE_ASYNC_LDS
  __attribute__((address_space(1))) v4i* gp =
      (__attribute__((address_space(1))) v4i*)g;
  __attribute__((address_space(3))) v4i* lp =
      (__attribute__((address_space(3))) v4i*)lds;
  for (int i = tid; i < n; i += nthr)
    __builtin_amdgcn_global_load_async_to_lds_b128(gp + i, lp + i, 0, 0);
  __builtin_amdgcn_s_wait_asynccnt(0);
#else
  const uint4* gp = (const uint4*)g;
  uint4* lp = (uint4*)lds;
  for (int i = tid; i < n; i += nthr) lp[i] = gp[i];
#endif
}

// ---------------------------------------------------------------------------
// degree-1 open B-spline basis, K=5, D=3 : 8 corners, weight + flat index
// ---------------------------------------------------------------------------
__device__ __forceinline__ void spline_basis(const float* __restrict__ u,
                                             float* basis, int* wi) {
  float f[3]; int i0[3];
#pragma unroll
  for (int d = 0; d < 3; ++d) {
    float v  = u[d] * 4.0f;               // (K-1)=4
    float fl = fminf(fmaxf(floorf(v), 0.0f), 3.0f);
    i0[d] = (int)fl;
    f[d]  = v - fl;
  }
#pragma unroll
  for (int s = 0; s < 8; ++s) {
    float w = 1.0f; int idx = 0, stride = 1;
#pragma unroll
    for (int d = 0; d < 3; ++d) {
      int bit = (s >> d) & 1;
      w   *= bit ? f[d] : (1.0f - f[d]);
      idx += (i0[d] + bit) * stride;
      stride *= 5;
    }
    basis[s] = w; wi[s] = idx;
  }
}

__device__ __forceinline__ float elu1(float x) {
  return x > 0.0f ? x : (expf(x) - 1.0f);
}

// ---------------------------------------------------------------------------
__global__ void k_zero(float* p, int n) {
  int i = blockIdx.x * blockDim.x + threadIdx.x;
  if (i < n) p[i] = 0.0f;
}

__global__ void k_izero(int* p, int n) {
  int i = blockIdx.x * blockDim.x + threadIdx.x;
  if (i < n) p[i] = 0;
}

// in-degree reciprocals (topology shared across batch)
__global__ void k_cnt(const int* __restrict__ ei, float* __restrict__ rcnt) {
  int n = threadIdx.x;
  if (n < NN) {
    int c = 0;
    for (int e = 0; e < EE; ++e) c += (ei[EE + e] == n);
    rcnt[n] = 1.0f / (float)(c > 0 ? c : 1);
  }
}

// ---------------------------------------------------------------------------
// Layer-1 edge messages: wave per (b,e), lane per output channel (F1=32)
// ---------------------------------------------------------------------------
__global__ void k_l1(const float* __restrict__ x, const int* __restrict__ ei,
                     const float* __restrict__ u, const float* __restrict__ W1,
                     float* __restrict__ h1buf) {
  int wave = blockIdx.x * 8 + (threadIdx.x >> 5);
  int lane = threadIdx.x & 31;
  int b = wave / EE, e = wave % EE;
  float basis[8]; int wi[8];
  spline_basis(u + (b * EE + e) * 3, basis, wi);
  int src = ei[e], dst = ei[EE + e];
  const float* xp = x + (b * NN + src) * 3;
  float x0 = xp[0], x1 = xp[1], x2 = xp[2];
  float acc = 0.0f;
#pragma unroll
  for (int s = 0; s < 8; ++s) {
    const float* Wp = W1 + wi[s] * 3 * F1;
    acc += basis[s] * (x0 * Wp[lane] + x1 * Wp[F1 + lane] + x2 * Wp[2 * F1 + lane]);
  }
  atomicAdd(&h1buf[(b * NN + dst) * F1 + lane], acc);
}

// Layer-1 finalize: mean + root + bias + ELU (in place)
__global__ void k_l1post(const float* __restrict__ x, const float* __restrict__ R1,
                         const float* __restrict__ b1, const float* __restrict__ rcnt,
                         float* __restrict__ h1buf) {
  int idx = blockIdx.x * blockDim.x + threadIdx.x;   // B*NN*F1 threads
  int o = idx & 31, n = (idx >> 5) % NN, b = idx / (NN * F1);
  const float* xp = x + (b * NN + n) * 3;
  float root = xp[0] * R1[o] + xp[1] * R1[F1 + o] + xp[2] * R1[2 * F1 + o];
  float h = h1buf[idx] * rcnt[n] + root + b1[o];
  h1buf[idx] = elu1(h);
}

// ---------------------------------------------------------------------------
// root2 = h1 @ R2 via v_wmma_f32_16x16x32_bf16 (wave per 16x16 tile, K=32).
// ISA §7.12.2 wave32 packing:
//   A: lane<16: row=lane,  K={0..7,16..23}; lane>=16: row=lane-16, K={8..15,24..31}
//   B: lanes 0-15: K={0..7,16..23}, N=lane; lanes 16-31: K={8..15,24..31}, N=lane-16
//   D: VGPR j: lanes 0-15 -> M=j, lanes 16-31 -> M=j+8; N=lane%16
// ---------------------------------------------------------------------------
__global__ void k_root2(const float* __restrict__ h1, const float* __restrict__ R2,
                        float* __restrict__ root2) {
  int tile = blockIdx.x * 8 + (threadIdx.x >> 5);   // 1536 tiles = 384 M x 4 N
  int lane = threadIdx.x & 31;
  int mt = tile >> 2, nt = tile & 3;
  int m0 = mt * 16, n0 = nt * 16;
  int half = lane >> 4, lm = lane & 15;
  int row = m0 + lm, col = n0 + lm;

  v16bf a, bf;
#pragma unroll
  for (int t = 0; t < 16; ++t) {
    int k = (t < 8) ? (half * 8 + t) : (16 + half * 8 + (t - 8));
    a[t] = (__bf16)h1[row * F1 + k];
  }
#pragma unroll
  for (int t = 0; t < 16; ++t) {
    int k = (t < 8) ? (half * 8 + t) : (16 + half * 8 + (t - 8));
    bf[t] = (__bf16)R2[k * F2 + col];
  }
  v8f c = {};
  c = __builtin_amdgcn_wmma_f32_16x16x32_bf16(false, a, false, bf,
                                              (short)0, c, false, false);
#pragma unroll
  for (int j = 0; j < 8; ++j) {
    int M = j + half * 8;
    root2[(m0 + M) * F2 + n0 + lm] = c[j];
  }
}

// ===========================================================================
// Layer-2 message pass, grouped-by-cell WMMA path
// ===========================================================================

// W2 f32 -> f16
__global__ void k_cvtw2(const float* __restrict__ w, _Float16* __restrict__ wh, int n) {
  int i = blockIdx.x * blockDim.x + threadIdx.x;
  if (i < n) wh[i] = (_Float16)w[i];
}

// histogram of cell usage: thread per (b,e), 8 counter atomics
__global__ void k_bincnt(const float* __restrict__ u, int* __restrict__ counts) {
  int i = blockIdx.x * blockDim.x + threadIdx.x;   // NEDGE threads
  float basis[8]; int wi[8];
  spline_basis(u + i * 3, basis, wi);
#pragma unroll
  for (int s = 0; s < 8; ++s) atomicAdd(&counts[wi[s]], 1);
}

// serial prefix scan over 125 bins (tiny): row offsets, tile offsets, cursors
__global__ void k_scan(const int* __restrict__ counts, int* __restrict__ binOff,
                       int* __restrict__ tileOff, int* __restrict__ cursor,
                       int* __restrict__ total) {
  if (threadIdx.x == 0) {
    int acc = 0, tacc = 0;
    for (int k = 0; k < KTOT; ++k) {
      binOff[k] = acc; tileOff[k] = tacc; cursor[k] = acc;
      int c = counts[k];
      acc  += c;
      tacc += (c + 15) >> 4;
    }
    binOff[KTOT] = acc; tileOff[KTOT] = tacc; total[0] = tacc;
  }
}

// scatter: wave per (b,e); lane holds feature `lane` of h1[src].
__global__ void k_scatter(const float* __restrict__ h1, const int* __restrict__ ei,
                          const float* __restrict__ u, int* __restrict__ cursor,
                          _Float16* __restrict__ arow, int* __restrict__ meta) {
  int wave = blockIdx.x * 8 + (threadIdx.x >> 5);
  int lane = threadIdx.x & 31;
  int b = wave / EE, e = wave % EE;
  float basis[8]; int wi[8];
  spline_basis(u + (b * EE + e) * 3, basis, wi);
  int src = ei[e], dst = ei[EE + e];
  float hval = h1[(b * NN + src) * F1 + lane];
#pragma unroll
  for (int s = 0; s < 8; ++s) {
    int pos = 0;
    if (lane == 0) pos = atomicAdd(&cursor[wi[s]], 1);
    pos = __shfl(pos, 0, 32);
    arow[(size_t)pos * F1 + lane] = (_Float16)(basis[s] * hval);
    if (lane == 0) meta[pos] = b * NN + dst;
  }
}

// Grouped GEMM: block = 4 waves = one ragged 16-row M tile x all 64 N cols of
// one bin; B slice (32x64 f16, 4KB) async-staged in LDS; D rows scattered with
// f32 global atomics via per-row metadata. v_wmma_f32_16x16x32_f16.
__global__ void k_mm(const _Float16* __restrict__ arow, const int* __restrict__ meta,
                     const _Float16* __restrict__ w2h, const int* __restrict__ binOff,
                     const int* __restrict__ tileOff, const int* __restrict__ total,
                     float* __restrict__ h2buf) {
  __shared__ _Float16 sB[F1 * F2];   // 4KB
  __shared__ int sMeta[16];
  int t = blockIdx.x;
  if (t >= total[0]) return;                         // uniform: safe with barrier
  // bin = upper_bound(tileOff, t) - 1
  int lo = 0, hi = KTOT;
  while (hi - lo > 1) { int mid = (lo + hi) >> 1; if (tileOff[mid] <= t) lo = mid; else hi = mid; }
  int bin      = lo;
  int rowStart = binOff[bin] + ((t - tileOff[bin]) << 4);
  int nrows    = binOff[bin + 1] - rowStart; if (nrows > 16) nrows = 16;

  int tid = threadIdx.x;
  stage_lds(sB, w2h + (size_t)bin * (F1 * F2), F1 * F2 * 2, tid, 128);
  if (tid < 16) sMeta[tid] = (tid < nrows) ? meta[rowStart + tid] : -1;
  __syncthreads();

  int wave = tid >> 5, lane = tid & 31;
  int n0 = wave * 16;
  int half = lane >> 4, lm = lane & 15;
  bool av = lm < nrows;
  const _Float16* ap = arow + (size_t)(rowStart + lm) * F1;

  v16h a, bh;
#pragma unroll
  for (int k = 0; k < 16; ++k) {
    int kk = (k < 8) ? (half * 8 + k) : (16 + half * 8 + (k - 8));
    a[k] = av ? ap[kk] : (_Float16)0.0f;             // zero-pad ragged rows
  }
#pragma unroll
  for (int k = 0; k < 16; ++k) {
    int kk = (k < 8) ? (half * 8 + k) : (16 + half * 8 + (k - 8));
    bh[k] = sB[kk * F2 + n0 + lm];
  }
  v8f c = {};
  c = __builtin_amdgcn_wmma_f32_16x16x32_f16(false, a, false, bh,
                                             (short)0, c, false, false);
#pragma unroll
  for (int j = 0; j < 8; ++j) {
    int M = j + half * 8;
    int m = sMeta[M];
    if (m >= 0) atomicAdd(&h2buf[m * F2 + n0 + lm], c[j]);
  }
}

// ---------------------------------------------------------------------------
// Fallback layer-2 edge messages (VALU, used only if workspace is too small)
// ---------------------------------------------------------------------------
__global__ void k_l2(const float* __restrict__ h1, const int* __restrict__ ei,
                     const float* __restrict__ u, const float* __restrict__ W2,
                     float* __restrict__ h2buf) {
  int wave = blockIdx.x * 8 + (threadIdx.x >> 5);
  int lane = threadIdx.x & 31;
  int b = wave / EE, e = wave % EE;
  float basis[8]; int wi[8];
  spline_basis(u + (b * EE + e) * 3, basis, wi);
  int src = ei[e], dst = ei[EE + e];
  float hval = h1[(b * NN + src) * F1 + lane];
  float acc0 = 0.0f, acc1 = 0.0f;
#pragma unroll
  for (int s = 0; s < 8; ++s) {
    const float* Wp = W2 + wi[s] * (F1 * F2);
    __builtin_prefetch(Wp, 0, 0);                    // global_prefetch_b8
    float p0 = 0.0f, p1 = 0.0f;
#pragma unroll
    for (int i = 0; i < F1; ++i) {
      float hv = __shfl(hval, i, 32);
      p0 += hv * Wp[i * F2 + lane];
      p1 += hv * Wp[i * F2 + 32 + lane];
    }
    acc0 += basis[s] * p0;
    acc1 += basis[s] * p1;
  }
  float* dp = &h2buf[(b * NN + dst) * F2 + lane];
  atomicAdd(dp, acc0);
  atomicAdd(dp + 32, acc1);
}

// Layer-2 finalize: mean + root2 + bias + ELU, write bf16 activations
__global__ void k_l2post(const float* __restrict__ h2buf, const float* __restrict__ root2,
                         const float* __restrict__ b2, const float* __restrict__ rcnt,
                         __bf16* __restrict__ h2bf) {
  int idx = blockIdx.x * blockDim.x + threadIdx.x;   // B*NN*F2 threads
  int o = idx & 63, n = (idx >> 6) % NN;
  float h = h2buf[idx] * rcnt[n] + root2[idx] + b2[o];
  h2bf[idx] = (__bf16)elu1(h);                       // idx == b*768 + n*64 + o
}

// fc1 weight fp32 -> bf16
__global__ void k_cvtw(const float* __restrict__ w, __bf16* __restrict__ wb, int n) {
  int i = blockIdx.x * blockDim.x + threadIdx.x;
  if (i < n) wb[i] = (__bf16)w[i];
}

// ---------------------------------------------------------------------------
// fc1: [512,768]bf16 @ [768,128]bf16 + bias, f32 accum via WMMA.
// Block = 8 waves; block owns 16 rows (A tile async-staged once in LDS, 24KB),
// wave w owns N tile w (16 cols). K loop: 24 steps of 32.
// ---------------------------------------------------------------------------
__global__ void k_fc1(const __bf16* __restrict__ A, const __bf16* __restrict__ W,
                      const float* __restrict__ bias, float* __restrict__ out) {
  __shared__ __bf16 sA[16 * FLAT];                   // 24KB
  int tid  = threadIdx.x;
  int wave = tid >> 5;
  int lane = tid & 31;
  int m0 = blockIdx.x * 16, n0 = wave * 16;
  int half = lane >> 4, lm = lane & 15;
  int col = n0 + lm;

  stage_lds(sA, A + (size_t)m0 * FLAT, 16 * FLAT * 2, tid, 256);
  __syncthreads();

  v8f c = {};
  for (int kk = 0; kk < FLAT; kk += 32) {
    v16bf a, bf;
#pragma unroll
    for (int t = 0; t < 16; ++t) {
      int k = kk + ((t < 8) ? (half * 8 + t) : (16 + half * 8 + (t - 8)));
      a[t] = sA[lm * FLAT + k];
    }
#pragma unroll
    for (int t = 0; t < 16; ++t) {
      int k = kk + ((t < 8) ? (half * 8 + t) : (16 + half * 8 + (t - 8)));
      bf[t] = W[k * FH + col];
    }
    c = __builtin_amdgcn_wmma_f32_16x16x32_bf16(false, a, false, bf,
                                                (short)0, c, false, false);
  }
#pragma unroll
  for (int j = 0; j < 8; ++j) {
    int M = j + half * 8;
    out[(m0 + M) * FH + n0 + lm] = c[j] + bias[n0 + lm];
  }
}

// fc2 + log_softmax (2 classes)
__global__ void k_fc2(const float* __restrict__ fin, const float* __restrict__ w,
                      const float* __restrict__ b, float* __restrict__ out) {
  int i = blockIdx.x * blockDim.x + threadIdx.x;
  if (i >= BB) return;
  float z0 = b[0], z1 = b[1];
  const float* f = fin + i * FH;
  for (int k = 0; k < FH; ++k) {
    float v = f[k];
    z0 += v * w[k * 2 + 0];
    z1 += v * w[k * 2 + 1];
  }
  float m = fmaxf(z0, z1);
  float lse = m + logf(expf(z0 - m) + expf(z1 - m));
  out[i * 2 + 0] = z0 - lse;
  out[i * 2 + 1] = z1 - lse;
}

// ---------------------------------------------------------------------------
extern "C" void kernel_launch(void* const* d_in, const int* in_sizes, int n_in,
                              void* d_out, int out_size, void* d_ws, size_t ws_size,
                              hipStream_t stream) {
  const float* x    = (const float*)d_in[0];
  const int*   ei   = (const int*)  d_in[1];
  const float* u    = (const float*)d_in[2];
  const float* W1   = (const float*)d_in[3];
  const float* R1   = (const float*)d_in[4];
  const float* b1   = (const float*)d_in[5];
  const float* W2   = (const float*)d_in[6];
  const float* R2   = (const float*)d_in[7];
  const float* b2   = (const float*)d_in[8];
  const float* fc1w = (const float*)d_in[9];
  const float* fc1b = (const float*)d_in[10];
  const float* fc2w = (const float*)d_in[11];
  const float* fc2b = (const float*)d_in[12];
  float* out = (float*)d_out;

  char* ws = (char*)d_ws;
  // ---- workspace layout (bytes, 256-aligned) ----
  float*    h1buf  = (float*)   (ws + 0);          // 786432
  float*    h2buf  = (float*)   (ws + 786432);     // 1572864
  float*    root2  = (float*)   (ws + 2359296);    // 1572864
  float*    rcnt   = (float*)   (ws + 3932160);    // 256
  __bf16*   h2bf   = (__bf16*)  (ws + 3932416);    // 786432
  __bf16*   wbf    = (__bf16*)  (ws + 4718848);    // 196608
  float*    fc1o   = (float*)   (ws + 4915456);    // 262144
  _Float16* w2h    = (_Float16*)(ws + 5177600);    // 512000
  int*      counts = (int*)     (ws + 5689600);    // 512
  int*      binOff = (int*)     (ws + 5690112);    // 512
  int*      tileOff= (int*)     (ws + 5690624);    // 512
  int*      cursor = (int*)     (ws + 5691136);    // 512
  int*      total  = (int*)     (ws + 5691648);    // 256
  int*      meta   = (int*)     (ws + 5691904);    // 786432
  _Float16* arow   = (_Float16*)(ws + 6478336);    // 12582912
  const size_t WS_NEED = 19061248;                 // end of arow

  // zero atomic accumulators (h1buf+h2buf contiguous: 589824 floats)
  k_zero<<<2304, 256, 0, stream>>>(h1buf, 589824);
  k_cnt <<<1, 32, 0, stream>>>(ei, rcnt);

  k_l1    <<<3072, 256, 0, stream>>>(x, ei, u, W1, h1buf);
  k_l1post<<<768,  256, 0, stream>>>(x, R1, b1, rcnt, h1buf);
  k_root2 <<<192,  256, 0, stream>>>(h1buf, R2, root2);

  if (ws_size >= WS_NEED) {
    // grouped-by-cell WMMA message pass
    k_cvtw2  <<<1000, 256, 0, stream>>>(W2, w2h, KTOT * F1 * F2);
    k_izero  <<<1, 256, 0, stream>>>(counts, KTOT);
    k_bincnt <<<96, 256, 0, stream>>>(u, counts);             // 24576 threads
    k_scan   <<<1, 32, 0, stream>>>(counts, binOff, tileOff, cursor, total);
    k_scatter<<<3072, 256, 0, stream>>>(h1buf, ei, u, cursor, arow, meta);
    k_mm     <<<MAXTILES, 128, 0, stream>>>(arow, meta, w2h, binOff, tileOff,
                                            total, h2buf);
  } else {
    // VALU fallback (weight-gather streaming)
    k_l2   <<<3072, 256, 0, stream>>>(h1buf, ei, u, W2, h2buf);
  }

  k_l2post<<<1536, 256, 0, stream>>>(h2buf, root2, b2, rcnt, h2bf);
  k_cvtw  <<<384,  256, 0, stream>>>(fc1w, wbf, FLAT * FH);
  k_fc1   <<<32,   256, 0, stream>>>(h2bf, wbf, fc1b, fc1o);
  k_fc2   <<<2,    256, 0, stream>>>(fc1o, fc2w, fc2b, out);
}
